// MoeRegGPT_15831249453473
// MI455X (gfx1250) — compile-verified, matrix operand order checked by reading, not run on hardware
//
#include <hip/hip_runtime.h>
#include <hip/hip_bf16.h>

// Problem constants
#define B_   2
#define S_   1024
#define T_   2048      // B*S tokens
#define D_   768
#define H_   12
#define DH_  64
#define V_   32000
#define L_   2
#define E_   8
#define F_   3072
#define EPS_ 1e-6f

typedef __attribute__((ext_vector_type(16))) __bf16 bf16x16;
typedef __attribute__((ext_vector_type(8)))  __bf16 bf16x8;
typedef __attribute__((ext_vector_type(8)))  float  v8f;

// ---------------------------------------------------------------------------
// Load a 16x32 (MxK) bf16 fragment from LDS in the CDNA5 wave32 WMMA layout.
// Lane l: row m = l&15, K-half = l>>4. VGPRs 0..3 hold K = kh*8 .. kh*8+7,
// VGPRs 4..7 hold K = 16+kh*8 .. 16+kh*8+7 (2 bf16 per VGPR). Both runs are
// contiguous 16B in a row-major LDS tile, so two b128 ds loads per fragment.
// The same loader serves B fragments when the B tile is staged transposed
// (LDS holds B^T row-major: row = n, cols = k).
// ---------------------------------------------------------------------------
__device__ __forceinline__ bf16x16 load_frag(const __bf16* base, int stride) {
  int lane = threadIdx.x & 31;
  const __bf16* p = base + (lane & 15) * stride + ((lane >> 4) << 3);
  bf16x8 lo = *(const bf16x8*)__builtin_assume_aligned(p, 16);
  bf16x8 hi = *(const bf16x8*)__builtin_assume_aligned(p + 16, 16);
  return __builtin_shufflevector(lo, hi, 0,1,2,3,4,5,6,7,8,9,10,11,12,13,14,15);
}

__device__ __forceinline__ float gelu_tanh(float x) {
  float x3 = x * x * x;
  return 0.5f * x * (1.0f + tanhf(0.7978845608028654f * (x + 0.044715f * x3)));
}

// ---------------------------------------------------------------------------
// Embedding gather: h[t, :] = embed[x[t], :]
// ---------------------------------------------------------------------------
__global__ __launch_bounds__(256) void embed_k(const int* __restrict__ x,
                                               const float* __restrict__ emb,
                                               float* __restrict__ h) {
  int i = blockIdx.x * 256 + threadIdx.x;
  if (i >= T_ * D_) return;
  int t = i / D_, d = i - t * D_;
  h[i] = emb[(size_t)x[t] * D_ + d];
}

// ---------------------------------------------------------------------------
// RMSNorm: one workgroup per token row.
// ---------------------------------------------------------------------------
__global__ __launch_bounds__(256) void rmsnorm_k(const float* __restrict__ h,
                                                 const float* __restrict__ w,
                                                 float* __restrict__ xn) {
  __shared__ float red[256];
  int t = blockIdx.x, tid = threadIdx.x;
  const float* row = h + (size_t)t * D_;
  float s = 0.f;
  for (int d = tid; d < D_; d += 256) { float v = row[d]; s += v * v; }
  red[tid] = s;
  __syncthreads();
  for (int off = 128; off > 0; off >>= 1) {
    if (tid < off) red[tid] += red[tid + off];
    __syncthreads();
  }
  float r = rsqrtf(red[0] / (float)D_ + EPS_);
  for (int d = tid; d < D_; d += 256)
    xn[(size_t)t * D_ + d] = row[d] * r * w[d];
}

// ---------------------------------------------------------------------------
// RoPE applied in-place to q and k. pair index p in [0,32) pairs (p, p+32).
// ---------------------------------------------------------------------------
__global__ __launch_bounds__(256) void rope_k(float* __restrict__ q,
                                              float* __restrict__ k) {
  int i = blockIdx.x * 256 + threadIdx.x;
  const int half = DH_ / 2;
  if (i >= T_ * H_ * half) return;
  int p = i % half;
  int r = i / half;
  int hh = r % H_;
  int t = r / H_;
  int s = t & (S_ - 1);
  float freq = __powf(10000.0f, -(float)p / (float)half);
  float ang = (float)s * freq;
  float c = __cosf(ang), sn = __sinf(ang);
  size_t base = (size_t)t * D_ + (size_t)hh * DH_;
  float q1 = q[base + p], q2 = q[base + p + half];
  q[base + p]        = q1 * c - q2 * sn;
  q[base + p + half] = q1 * sn + q2 * c;
  float k1 = k[base + p], k2 = k[base + p + half];
  k[base + p]        = k1 * c - k2 * sn;
  k[base + p + half] = k1 * sn + k2 * c;
}

// ---------------------------------------------------------------------------
// Router: one wave per token. logits = xn @ rw (768x8), softmax, top-1.
// Buckets tokens per expert via atomics (order nondeterministic, result set
// per expert is deterministic; downstream GEMM is order-invariant).
// ---------------------------------------------------------------------------
__global__ __launch_bounds__(256) void router_k(const float* __restrict__ xn,
                                                const float* __restrict__ rw,
                                                int* __restrict__ cnt,
                                                int* __restrict__ tokidx,
                                                float* __restrict__ gate) {
  int wave = threadIdx.x >> 5, lane = threadIdx.x & 31;
  int t = blockIdx.x * 8 + wave;
  if (t >= T_) return;
  float acc[E_];
#pragma unroll
  for (int e = 0; e < E_; ++e) acc[e] = 0.f;
  for (int d = lane; d < D_; d += 32) {
    float xv = xn[(size_t)t * D_ + d];
#pragma unroll
    for (int e = 0; e < E_; ++e) acc[e] += xv * rw[(size_t)d * E_ + e];
  }
#pragma unroll
  for (int e = 0; e < E_; ++e)
    for (int m = 16; m > 0; m >>= 1) acc[e] += __shfl_xor(acc[e], m, 32);
  if (lane == 0) {
    float mx = acc[0];
    int im = 0;
    for (int e = 1; e < E_; ++e)
      if (acc[e] > mx) { mx = acc[e]; im = e; }
    float ssum = 0.f;
    for (int e = 0; e < E_; ++e) ssum += __expf(acc[e] - mx);
    float g = 1.0f / ssum;  // softmax value at the argmax
    int pos = atomicAdd(&cnt[im], 1);
    tokidx[im * T_ + pos] = t;
    gate[t] = g;
  }
}

// ---------------------------------------------------------------------------
// Generic tiled WMMA GEMM:  C[m,:] (+)= act(A[m,:] @ B) [* gate] [+ bias]
// Block tile 128x64, K-step 32, 8 waves each own a 16x64 strip (4 fragments).
// Software-pipelined: the next K-tile's global loads are issued into registers
// before the current tile's WMMAs, so HBM latency overlaps matrix compute;
// the implicit s_wait_loadcnt lands at the register->LDS commit one iteration
// later. Wave 0 additionally issues global_prefetch for the B tile two steps
// ahead (L2 warm-up; proj_w / expert tiles are L2-resident on re-read).
// GATHER: blockIdx.z = expert; A rows gathered / C rows scattered via tokidx,
//         row count from cnt[e], B offset e*K*N.
// ---------------------------------------------------------------------------
template <bool GATHER, bool ACC, bool GELU_, bool BIAS, bool SCALE>
__global__ __launch_bounds__(256) void gemm_k(
    const float* __restrict__ A, int lda, const float* __restrict__ Bw, int N,
    int K, float* __restrict__ C, int ldc, int M, const int* __restrict__ cnt,
    const int* __restrict__ tokidx, const float* __restrict__ gate,
    const float* __restrict__ bias) {
  int e = GATHER ? blockIdx.z : 0;
  int Meff = GATHER ? cnt[e] : M;
  int m0 = blockIdx.x * 128;
  if (m0 >= Meff) return;  // uniform across the workgroup
  int n0 = blockIdx.y * 64;
  const float* Bp = Bw + (size_t)e * (size_t)K * (size_t)N;
  const int* tix = GATHER ? (tokidx + e * T_) : nullptr;

  __shared__ __bf16 sA[128][40];  // row-major MxK tile (stride 40: 16B aligned)
  __shared__ __bf16 sB[64][40];   // B^T tile: row = n, cols = k

  int tid = threadIdx.x, wave = tid >> 5, lane = tid & 31;

  v8f c[4];
#pragma unroll
  for (int i = 0; i < 4; ++i)
#pragma unroll
    for (int j = 0; j < 8; ++j) c[i][j] = 0.f;

  float ra[16];  // register staging: A 128x32 / 256 threads
  float rb[8];   // register staging: B 32x64 / 256 threads

  auto fetch = [&](int kk0) {
#pragma unroll
    for (int i = 0; i < 16; ++i) {
      int idx = tid + i * 256;
      int r = idx >> 5, kk = idx & 31;
      int row = m0 + r;
      float v = 0.f;
      if (row < Meff) {
        int ar = GATHER ? tix[row] : row;
        v = A[(size_t)ar * lda + kk0 + kk];
      }
      ra[i] = v;
    }
#pragma unroll
    for (int i = 0; i < 8; ++i) {
      int idx = tid + i * 256;
      int kk = idx >> 6, nn = idx & 63;
      int col = n0 + nn;
      rb[i] = (col < N) ? Bp[(size_t)(kk0 + kk) * N + col] : 0.f;
    }
  };
  auto commit = [&]() {
#pragma unroll
    for (int i = 0; i < 16; ++i) {
      int idx = tid + i * 256;
      sA[idx >> 5][idx & 31] = (__bf16)ra[i];
    }
#pragma unroll
    for (int i = 0; i < 8; ++i) {
      int idx = tid + i * 256;
      sB[idx & 63][idx >> 6] = (__bf16)rb[i];
    }
  };

  fetch(0);
  for (int k0 = 0; k0 < K; k0 += 32) {
    commit();
    __syncthreads();
    if (k0 + 32 < K) fetch(k0 + 32);      // in-flight during WMMAs below
    if (wave == 0 && k0 + 64 < K) {       // warm L2 two tiles ahead
      const float* pf = &Bp[(size_t)(k0 + 64 + lane) * N + n0];
      __builtin_prefetch(pf, 0, 1);
      __builtin_prefetch(pf + 32, 0, 1);  // cover the 256B row segment
    }
    bf16x16 a = load_frag(&sA[wave * 16][0], 40);
#pragma unroll
    for (int nb = 0; nb < 4; ++nb) {
      bf16x16 b = load_frag(&sB[nb * 16][0], 40);
      c[nb] = __builtin_amdgcn_wmma_f32_16x16x32_bf16(false, a, false, b,
                                                      (short)0, c[nb], false,
                                                      false);
    }
    __syncthreads();
  }

  // Epilogue: C/D layout -> lane holds col = n0+nb*16+(lane&15),
  // rows m0 + wave*16 + (lane>>4)*8 + r for r in 0..7.
  int rbase = m0 + wave * 16 + ((lane >> 4) << 3);
  int cbase = n0 + (lane & 15);
#pragma unroll
  for (int nb = 0; nb < 4; ++nb) {
    int col = cbase + nb * 16;
    if (col >= N) continue;
#pragma unroll
    for (int r = 0; r < 8; ++r) {
      int row = rbase + r;
      if (row >= Meff) continue;
      float v = c[nb][r];
      if (BIAS) v += bias[col];
      if (GELU_) v = gelu_tanh(v);
      int crow = GATHER ? tix[row] : row;
      if (SCALE) v *= gate[crow];
      float* dst = C + (size_t)crow * ldc + col;
      if (ACC) *dst += v; else *dst = v;
    }
  }
}

// ---------------------------------------------------------------------------
// Flash attention (causal): grid = (S/128, H, B), 8 waves x 16 q-rows each.
// Scale 1/sqrt(DH) folded into Q at staging. Online softmax per row with
// 16-lane butterfly reductions (row stats live per half-wave). P converted
// from C-layout to A-layout through LDS before the P@V WMMA.
// ---------------------------------------------------------------------------
__global__ __launch_bounds__(256) void flash_k(const float* __restrict__ q,
                                               const float* __restrict__ k,
                                               const float* __restrict__ v,
                                               float* __restrict__ o) {
  __shared__ __bf16 sQ[128][72];  // 128 q-rows x 64 dh (stride 72: 16B aligned)
  __shared__ __bf16 sK[32][72];   // 32 keys x 64 dh (== B^T for q.k^T)
  __shared__ __bf16 sV[64][40];   // V^T: dh x 32 keys (== B^T for p.v)
  __shared__ __bf16 sP[128][40];  // per-wave 16x32 probability tiles

  int tid = threadIdx.x, wave = tid >> 5, lane = tid & 31;
  int q0 = blockIdx.x * 128;
  int hh = blockIdx.y;
  int bb = blockIdx.z;
  const size_t hoff = (size_t)hh * DH_;

  // Stage Q tile (pre-scaled by 1/sqrt(DH) = 0.125)
#pragma unroll
  for (int i = 0; i < 32; ++i) {
    int idx = tid + i * 256;
    int r = idx >> 6, dh = idx & 63;
    int t = bb * S_ + q0 + r;
    sQ[r][dh] = (__bf16)(q[(size_t)t * D_ + hoff + dh] * 0.125f);
  }
  __syncthreads();
  bf16x16 aq0 = load_frag(&sQ[wave * 16][0], 72);   // dh 0..31
  bf16x16 aq1 = load_frag(&sQ[wave * 16][32], 72);  // dh 32..63

  int qrow_lo = q0 + wave * 16;
  int qend = qrow_lo + 15;
  int myrow = qrow_lo + ((lane >> 4) << 3);

  float mrun[8], lrun[8];
  v8f oacc[4];
#pragma unroll
  for (int r = 0; r < 8; ++r) { mrun[r] = -1e30f; lrun[r] = 0.f; }
#pragma unroll
  for (int nb = 0; nb < 4; ++nb)
#pragma unroll
    for (int r = 0; r < 8; ++r) oacc[nb][r] = 0.f;

  int nblocks = (q0 + 128) / 32;  // causal: keys up to q0+127
  for (int jb = 0; jb < nblocks; ++jb) {
    int j0 = jb * 32;
    // Stage K tile and transposed V tile
#pragma unroll
    for (int i = 0; i < 8; ++i) {
      int idx = tid + i * 256;
      int key = idx >> 6, dh = idx & 63;
      int t = bb * S_ + j0 + key;
      sK[key][dh] = (__bf16)k[(size_t)t * D_ + hoff + dh];
      sV[dh][key] = (__bf16)v[(size_t)t * D_ + hoff + dh];
    }
    __syncthreads();

    bool active = (j0 <= qend);  // wave-uniform -> EXEC all-ones inside
    if (active) {
      bf16x16 b00 = load_frag(&sK[0][0], 72);
      bf16x16 b01 = load_frag(&sK[0][32], 72);
      bf16x16 b10 = load_frag(&sK[16][0], 72);
      bf16x16 b11 = load_frag(&sK[16][32], 72);
      v8f s0, s1;
#pragma unroll
      for (int r = 0; r < 8; ++r) { s0[r] = 0.f; s1[r] = 0.f; }
      s0 = __builtin_amdgcn_wmma_f32_16x16x32_bf16(false, aq0, false, b00,
                                                   (short)0, s0, false, false);
      s0 = __builtin_amdgcn_wmma_f32_16x16x32_bf16(false, aq1, false, b01,
                                                   (short)0, s0, false, false);
      s1 = __builtin_amdgcn_wmma_f32_16x16x32_bf16(false, aq0, false, b10,
                                                   (short)0, s1, false, false);
      s1 = __builtin_amdgcn_wmma_f32_16x16x32_bf16(false, aq1, false, b11,
                                                   (short)0, s1, false, false);

      int colb = j0 + (lane & 15);
      float mt[8];
#pragma unroll
      for (int r = 0; r < 8; ++r) {
        int row = myrow + r;
        float a0 = (colb <= row) ? s0[r] : -1e9f;
        float a1 = (colb + 16 <= row) ? s1[r] : -1e9f;
        s0[r] = a0; s1[r] = a1;
        mt[r] = fmaxf(a0, a1);
      }
#pragma unroll
      for (int r = 0; r < 8; ++r)
        for (int msk = 1; msk < 16; msk <<= 1)
          mt[r] = fmaxf(mt[r], __shfl_xor(mt[r], msk, 32));

      float alpha[8], psum[8];
#pragma unroll
      for (int r = 0; r < 8; ++r) {
        float mnew = fmaxf(mrun[r], mt[r]);
        alpha[r] = __expf(mrun[r] - mnew);
        mrun[r] = mnew;
        float p0 = __expf(s0[r] - mnew);
        float p1 = __expf(s1[r] - mnew);
        s0[r] = p0; s1[r] = p1;
        psum[r] = p0 + p1;
      }
#pragma unroll
      for (int r = 0; r < 8; ++r) {
        for (int msk = 1; msk < 16; msk <<= 1)
          psum[r] += __shfl_xor(psum[r], msk, 32);
        lrun[r] = lrun[r] * alpha[r] + psum[r];
      }
#pragma unroll
      for (int nb = 0; nb < 4; ++nb)
#pragma unroll
        for (int r = 0; r < 8; ++r) oacc[nb][r] *= alpha[r];

      // Write P (C-layout) into this wave's private LDS strip
      int prow = wave * 16 + ((lane >> 4) << 3);
      int pc = lane & 15;
#pragma unroll
      for (int r = 0; r < 8; ++r) {
        sP[prow + r][pc] = (__bf16)s0[r];
        sP[prow + r][pc + 16] = (__bf16)s1[r];
      }
    }
    __syncthreads();  // make sP visible for A-layout fragment reads
    if (active) {
      bf16x16 pA = load_frag(&sP[wave * 16][0], 40);
#pragma unroll
      for (int nb = 0; nb < 4; ++nb) {
        bf16x16 vb = load_frag(&sV[nb * 16][0], 40);
        oacc[nb] = __builtin_amdgcn_wmma_f32_16x16x32_bf16(
            false, pA, false, vb, (short)0, oacc[nb], false, false);
      }
    }
    __syncthreads();  // before restaging K/V
  }

#pragma unroll
  for (int r = 0; r < 8; ++r) {
    float inv = 1.0f / lrun[r];
    int row = myrow + r;
    size_t base = (size_t)(bb * S_ + row) * D_ + hoff;
#pragma unroll
    for (int nb = 0; nb < 4; ++nb)
      o[base + nb * 16 + (lane & 15)] = oacc[nb][r] * inv;
  }
}

// ---------------------------------------------------------------------------
// Host-side orchestration
// ---------------------------------------------------------------------------
extern "C" void kernel_launch(void* const* d_in, const int* in_sizes, int n_in,
                              void* d_out, int out_size, void* d_ws,
                              size_t ws_size, hipStream_t stream) {
  (void)in_sizes; (void)n_in; (void)out_size;

  const int*   x           = (const int*)d_in[0];
  const float* embed       = (const float*)d_in[1];
  const float* attn_norm_w = (const float*)d_in[2];
  const float* wq          = (const float*)d_in[3];
  const float* wk          = (const float*)d_in[4];
  const float* wv          = (const float*)d_in[5];
  const float* wo          = (const float*)d_in[6];
  const float* ffn_norm_w  = (const float*)d_in[7];
  const float* router_w    = (const float*)d_in[8];
  const float* w1          = (const float*)d_in[9];
  const float* w2          = (const float*)d_in[10];
  const float* proj_w      = (const float*)d_in[11];
  const float* proj_b      = (const float*)d_in[12];
  float*       out         = (float*)d_out;

  // Workspace layout
  const size_t nTD = (size_t)T_ * D_;
  const size_t nTF = (size_t)T_ * F_;
  float* h    = (float*)d_ws;
  float* xn   = h + nTD;
  float* qb   = xn + nTD;
  float* kb   = qb + nTD;
  float* vb   = kb + nTD;
  float* ao   = vb + nTD;
  float* h1   = ao + nTD;
  float* gate = h1 + nTF;
  int* tokidx = (int*)(gate + T_);
  int* cnt    = tokidx + E_ * T_;
  size_t need = (size_t)(cnt + E_ - (int*)d_ws) * sizeof(int);
  if (ws_size < need) return;

  // Embedding
  embed_k<<<(T_ * D_ + 255) / 256, 256, 0, stream>>>(x, embed, h);

  const dim3 gProj(T_ / 128, V_ / 64, 1);   // 16 x 500
  const dim3 gDD(T_ / 128, D_ / 64, 1);     // 16 x 12
  const dim3 gW1(T_ / 128, F_ / 64, E_);    // 16 x 48 x 8
  const dim3 gW2(T_ / 128, D_ / 64, E_);    // 16 x 12 x 8
  const dim3 gFlash(S_ / 128, H_, B_);      // 8 x 12 x 2

  for (int l = 0; l < L_; ++l) {
    // ---- Attention block ----
    rmsnorm_k<<<T_, 256, 0, stream>>>(h, attn_norm_w + (size_t)l * D_, xn);
    gemm_k<false, false, false, false, false><<<gDD, 256, 0, stream>>>(
        xn, D_, wq + (size_t)l * D_ * D_, D_, D_, qb, D_, T_, nullptr, nullptr,
        nullptr, nullptr);
    gemm_k<false, false, false, false, false><<<gDD, 256, 0, stream>>>(
        xn, D_, wk + (size_t)l * D_ * D_, D_, D_, kb, D_, T_, nullptr, nullptr,
        nullptr, nullptr);
    gemm_k<false, false, false, false, false><<<gDD, 256, 0, stream>>>(
        xn, D_, wv + (size_t)l * D_ * D_, D_, D_, vb, D_, T_, nullptr, nullptr,
        nullptr, nullptr);
    rope_k<<<(T_ * H_ * (DH_ / 2) + 255) / 256, 256, 0, stream>>>(qb, kb);
    flash_k<<<gFlash, 256, 0, stream>>>(qb, kb, vb, ao);
    gemm_k<false, true, false, false, false><<<gDD, 256, 0, stream>>>(
        ao, D_, wo + (size_t)l * D_ * D_, D_, D_, h, D_, T_, nullptr, nullptr,
        nullptr, nullptr);  // h += ao @ wo

    // ---- MoE block ----
    rmsnorm_k<<<T_, 256, 0, stream>>>(h, ffn_norm_w + (size_t)l * D_, xn);
    hipMemsetAsync(cnt, 0, E_ * sizeof(int), stream);
    router_k<<<T_ / 8, 256, 0, stream>>>(
        xn, router_w + (size_t)l * D_ * E_, cnt, tokidx, gate);
    // h1[t] = gelu(xn[t] @ w1[e(t)])  (gathered per expert)
    gemm_k<true, false, true, false, false><<<gW1, 256, 0, stream>>>(
        xn, D_, w1 + (size_t)l * E_ * D_ * F_, F_, D_, h1, F_, T_, cnt, tokidx,
        nullptr, nullptr);
    // h[t] += gate[t] * (h1[t] @ w2[e(t)])
    gemm_k<true, true, false, false, true><<<gW2, 256, 0, stream>>>(
        h1, F_, w2 + (size_t)l * E_ * F_ * D_, D_, F_, h, D_, T_, cnt, tokidx,
        gate, nullptr);
  }

  // Final projection: out = h @ proj_w + proj_b
  gemm_k<false, false, false, true, false><<<gProj, 256, 0, stream>>>(
      h, D_, proj_w, V_, D_, out, V_, T_, nullptr, nullptr, nullptr, proj_b);
}
